// BiLSTM_89086211653815
// MI455X (gfx1250) — matrix operand order repaired and checked by reading, not compiled, not run on hardware
//
#include <hip/hip_runtime.h>

// Problem constants (match reference): V=128, H=512, B=512, S=256
static constexpr int NV = 128;
static constexpr int NH = 512;
static constexpr int NB = 512;
static constexpr int NS = 256;

typedef __attribute__((ext_vector_type(8)))  float  v8f;
typedef __attribute__((ext_vector_type(8)))  __bf16 v8bf;
typedef __attribute__((ext_vector_type(16))) __bf16 v16bf;

// LDS: per-WG weight slab, gate-interleaved: [K=512][gate=4][72] bf16,
// row padded 288->296 elems (592 B: 16B aligned rows; gate offs fit DS imm)
static constexpr int    LDS_ROW   = 296;
static constexpr size_t LDS_BYTES = (size_t)NH * LDS_ROW * 2;  // 303104 <= 320K

// ---------------- workspace layout (bytes) ----------------
static constexpr size_t SZ_WHT  = 2ull * 4 * NH * NH * 2;   // bf16 WhT  [2][4][Hin][Hout]
static constexpr size_t SZ_WFCT = 2ull * NH * NV * 2;       // bf16 WfcT [2H][V]
static constexpr size_t SZ_C    = 2ull * NB * NH * 4;       // f32 cell state [2][B][H]
static constexpr size_t OFF_WHT  = 0;
static constexpr size_t OFF_WFCT = OFF_WHT + SZ_WHT;
static constexpr size_t OFF_C    = OFF_WFCT + SZ_WFCT;
static constexpr size_t OFF_BAR  = OFF_C + SZ_C;            // 256 B barrier counters
static constexpr size_t OFF_HS   = OFF_BAR + 256;           // bf16 hs [2][S][B][H]

// ---------------- helpers ----------------
static __device__ __forceinline__ v16bf cat16(v8bf lo, v8bf hi) {
  v16bf r;
#pragma unroll
  for (int i = 0; i < 8; ++i) { r[i] = lo[i]; r[i + 8] = hi[i]; }
  return r;
}

static __device__ __forceinline__ v8f wmma_bf16(v16bf a, v16bf b, v8f c) {
  // D = A(16x32 bf16) * B(32x16 bf16) + C(16x16 f32)
  return __builtin_amdgcn_wmma_f32_16x16x32_bf16(false, a, false, b, (short)0, c,
                                                 false, false);
}

static __device__ __forceinline__ float sigm(float x) {
  return 1.0f / (1.0f + __expf(-x));
}

// per-direction grid barrier (generation counter; counters zeroed by
// hipMemsetAsync before launch on every call)
static __device__ __forceinline__ void grid_barrier(unsigned* cnt, unsigned* gen,
                                                    unsigned nb) {
  __threadfence();
  __syncthreads();
  if (threadIdx.x == 0) {
    volatile unsigned* vgen = gen;
    unsigned g = *vgen;
    if (atomicAdd(cnt, 1u) == nb - 1u) {
      atomicExch(cnt, 0u);
      __threadfence();
      atomicAdd(gen, 1u);
    } else {
      while (*vgen == g) { __builtin_amdgcn_s_sleep(8); }
    }
  }
  __syncthreads();
  __threadfence();
}

// ---------------- prep: bf16 transpose of Wh (both dirs) and Wfc ----------------
__global__ void k_prep(const float* __restrict__ Wh_f, const float* __restrict__ Wh_b,
                       const float* __restrict__ Wfc,
                       __bf16* __restrict__ whT, __bf16* __restrict__ wfcT) {
  int idx = blockIdx.x * blockDim.x + threadIdx.x;
  if (idx < 4 * NH * NH) {                 // Wh: [g][kout][hin] -> WhT: [g][hin][kout]
    int g  = idx / (NH * NH);
    int rm = idx % (NH * NH);
    int ko = rm / NH, hi = rm % NH;
    size_t dst = ((size_t)g * NH + hi) * NH + ko;
    whT[dst]                       = (__bf16)Wh_f[idx];
    whT[(size_t)4 * NH * NH + dst] = (__bf16)Wh_b[idx];
  }
  if (idx < 2 * NH * NV) {                 // Wfc: [v][h] -> WfcT: [h][v]
    int h = idx / NV, v = idx % NV;
    wfcT[idx] = (__bf16)Wfc[(size_t)v * (2 * NH) + h];
  }
}

// ---------------- persistent bidirectional LSTM ----------------
// grid = (64, 2): 8x8 tiles of 64(batch) x 64(h_out) per direction, 1 WG/WGP.
// Entire recurrent-weight tile lives in LDS (gate-interleaved) for the whole
// sequence; the K-loop is barrier-free: ds_load + global_load(A) + wmma only.
__global__ __launch_bounds__(256) void k_bilstm(
    const int* __restrict__ x,
    const float* __restrict__ Wx_f, const float* __restrict__ bx_f,
    const float* __restrict__ bh_f,
    const float* __restrict__ Wx_b, const float* __restrict__ bx_b,
    const float* __restrict__ bh_b,
    const __bf16* __restrict__ whT,   // [2][4][Hin][Hout] bf16
    float* __restrict__ cst,          // [2][B][H] f32
    __bf16* __restrict__ hs,          // [2][S][B][H] bf16 (scan order)
    unsigned* __restrict__ bar) {
  extern __shared__ __bf16 ldsB[];   // [NH][4][72] (row = LDS_ROW elems)

  const int d     = blockIdx.y;
  const int tileM = (blockIdx.x & 7) * 64;   // batch tile
  const int tileN = (blockIdx.x >> 3) * 64;  // h_out tile
  const int tid = threadIdx.x, w = tid >> 5, l = tid & 31;
  const int wm = w & 1, wn = w >> 1;
  const int colL = l & 15;            // C/D col within 16
  const int hi8  = (l >> 4) * 8;      // C/D row offset for upper lane half
  const int kb   = hi8;               // A fragment K base per lane half
  const int ncol = tileN + wn * 16 + colL;   // this lane's global h_out column

  const float*  Wx   = d ? Wx_b : Wx_f;
  const float*  bx   = d ? bx_b : bx_f;
  const float*  bh   = d ? bh_b : bh_f;
  const __bf16* whTd = whT + (size_t)d * 4 * NH * NH;
  float*        cstd = cst + (size_t)d * NB * NH;
  __bf16*       hsd  = hs  + (size_t)d * NS * NB * NH;
  unsigned* cnt = bar + d * 2;
  unsigned* gen = bar + d * 2 + 1;

  // ---- one-time preload: 4x512x64 WhT tile -> LDS, gate-interleaved ----
  // 16384 b128 segments: idx -> k = idx>>5, g = (idx>>3)&3, n = (idx&7)*8
  for (int it = 0; it < 64; it += 4) {
    uint4 tmp[4];
#pragma unroll
    for (int j = 0; j < 4; ++j) {
      int idx = (it + j) * 256 + tid;
      int k = idx >> 5, g = (idx >> 3) & 3, n = (idx & 7) * 8;
      const __bf16* src = whTd + ((size_t)(g * NH + k)) * NH + tileN + n;
      tmp[j] = *(const uint4*)src;
      __builtin_prefetch(src + 32 * NH, 0, 1);
    }
#pragma unroll
    for (int j = 0; j < 4; ++j) {
      int idx = (it + j) * 256 + tid;
      int k = idx >> 5, g = (idx >> 3) & 3, n = (idx & 7) * 8;
      *(uint4*)&ldsB[(size_t)k * LDS_ROW + g * 72 + n] = tmp[j];
    }
  }
  __syncthreads();

  float bsum[4];
#pragma unroll
  for (int g = 0; g < 4; ++g) bsum[g] = bx[g * NH + ncol] + bh[g * NH + ncol];

  // per-lane invariant bases
  const __bf16* bwave = ldsB + (size_t)kb * LDS_ROW + wn * 16 + colL;  // B frag base
  const size_t  arowA0 = (size_t)(tileM + wm * 32 + colL) * NH + kb;  // mt=0 row
  unsigned* dummy = bar;  (void)dummy;

  for (int t = 0; t < NS; ++t) {
    const int p = d ? (NS - 1 - t) : t;  // backward dir scans reversed sequence

    // ---- init accumulators with one-hot(x)@Wx gather + biases ----
    int xv[2][8];
#pragma unroll
    for (int mt = 0; mt < 2; ++mt)
#pragma unroll
      for (int r = 0; r < 8; ++r) {
        int row = tileM + wm * 32 + mt * 16 + r + hi8;
        xv[mt][r] = x[row * NS + p];
      }
    v8f acc[4][2];
#pragma unroll
    for (int g = 0; g < 4; ++g) {
      const float* wxrow = Wx + ((size_t)g * NH + ncol) * NV;
#pragma unroll
      for (int mt = 0; mt < 2; ++mt)
#pragma unroll
        for (int r = 0; r < 8; ++r)
          acc[g][mt][r] = wxrow[xv[mt][r]] + bsum[g];
    }

    // ---- recurrent GEMM: g += h_{t-1} x WhT (K = 512, barrier-free) ----
    if (t > 0) {
      const __bf16* hprev = hsd + (size_t)(t - 1) * NB * NH;
      const __bf16* ap0 = hprev + arowA0;          // mt=0; mt=1 at +16*NH
#pragma unroll 4
      for (int kk = 0; kk < NH; kk += 32) {
        v16bf a[2];
#pragma unroll
        for (int mt = 0; mt < 2; ++mt) {
          // A 16x32 bf16: lane row = l&15; K = kb+{0..7} and kb+16+{0..7}
          const __bf16* ap = ap0 + mt * (16 * NH) + kk;
          a[mt] = cat16(*(const v8bf*)ap, *(const v8bf*)(ap + 16));
        }
        const __bf16* bbase = bwave + (size_t)kk * LDS_ROW;  // k rows = kk + kb + {0..7}, {16..23}
#pragma unroll
        for (int g = 0; g < 4; ++g) {
          // B 32x16 bf16: lane = N col (l&15); K = kb+{0..7} and kb+16+{0..7}
          const __bf16* bp = bbase + g * 72;
          v16bf bfr;
#pragma unroll
          for (int s = 0; s < 8; ++s) { bfr[s] = bp[s * LDS_ROW]; bfr[s + 8] = bp[(s + 16) * LDS_ROW]; }
          acc[g][0] = wmma_bf16(a[0], bfr, acc[g][0]);
          acc[g][1] = wmma_bf16(a[1], bfr, acc[g][1]);
        }
      }
    }

    // ---- fused LSTM pointwise epilogue ----
    __bf16* hout = hsd + (size_t)t * NB * NH;
#pragma unroll
    for (int mt = 0; mt < 2; ++mt)
#pragma unroll
      for (int r = 0; r < 8; ++r) {
        int row  = tileM + wm * 32 + mt * 16 + r + hi8;
        size_t ci = (size_t)row * NH + ncol;
        float gi = sigm(acc[0][mt][r]);
        float gf = sigm(acc[1][mt][r]);
        float go = sigm(acc[2][mt][r]);
        float gg = tanhf(acc[3][mt][r]);
        float cold = (t == 0) ? 0.0f : cstd[ci];
        float cn = gf * cold + gi * gg;
        cstd[ci] = cn;
        hout[ci] = (__bf16)(go * tanhf(cn));
      }

    grid_barrier(cnt, gen, gridDim.x);  // h_t visible device-wide before t+1
  }
}

// ---------------- final FC: y[b,s,:] = [hf|hb] @ WfcT + bfc ----------------
// grid = (B*S)/128 blocks; 8 waves each own one 16-row M tile x all 8 N tiles.
__global__ __launch_bounds__(256) void k_fc(const __bf16* __restrict__ hs,
                                            const __bf16* __restrict__ wfcT,
                                            const float* __restrict__ bfc,
                                            float* __restrict__ out) {
  const __bf16* hsf = hs;
  const __bf16* hsb = hs + (size_t)NS * NB * NH;
  const int tid = threadIdx.x, w = tid >> 5, l = tid & 31;
  const int colL = l & 15, hi8 = (l >> 4) * 8, kb = hi8;
  const int m0 = blockIdx.x * 128 + w * 16;  // flat row m = s*B + b

  v8f acc[8];
#pragma unroll
  for (int nt = 0; nt < 8; ++nt) {
    float bv = bfc[nt * 16 + colL];
#pragma unroll
    for (int r = 0; r < 8; ++r) acc[nt][r] = bv;
  }

  const int rowA = m0 + colL;           // A-lane row (rows of a tile share s)
  const int sA = rowA >> 9, bA = rowA & (NB - 1);
  const size_t arow_f = (size_t)rowA * NH;                        // hf[s][b]
  const size_t arow_b = ((size_t)(NS - 1 - sA) * NB + bA) * NH;   // hb scan rev

#pragma unroll 2
  for (int kk = 0; kk < 2 * NH; kk += 32) {
    const __bf16* ap = (kk < NH) ? (hsf + arow_f + kk + kb)
                                 : (hsb + arow_b + (kk - NH) + kb);
    v16bf a = cat16(*(const v8bf*)ap, *(const v8bf*)(ap + 16));
    const __bf16* bbase = wfcT + (size_t)(kk + kb) * NV + colL;
#pragma unroll
    for (int nt = 0; nt < 8; ++nt) {
      v16bf bfr;
#pragma unroll
      for (int s = 0; s < 8; ++s) { bfr[s] = bbase[s * NV + nt * 16]; bfr[s + 8] = bbase[(s + 16) * NV + nt * 16]; }
      acc[nt] = wmma_bf16(a, bfr, acc[nt]);
    }
  }

#pragma unroll
  for (int nt = 0; nt < 8; ++nt)
#pragma unroll
    for (int r = 0; r < 8; ++r) {
      int row = m0 + r + hi8;
      int s2 = row >> 9, b2 = row & (NB - 1);
      out[((size_t)b2 * NS + s2) * NV + nt * 16 + colL] = acc[nt][r];
    }
}

// ---------------- launch ----------------
extern "C" void kernel_launch(void* const* d_in, const int* in_sizes, int n_in,
                              void* d_out, int out_size, void* d_ws, size_t ws_size,
                              hipStream_t stream) {
  (void)in_sizes; (void)n_in; (void)out_size; (void)ws_size;
  const int*   x    = (const int*)d_in[0];
  const float* Wx_f = (const float*)d_in[1];
  const float* Wh_f = (const float*)d_in[2];
  const float* bx_f = (const float*)d_in[3];
  const float* bh_f = (const float*)d_in[4];
  const float* Wx_b = (const float*)d_in[5];
  const float* Wh_b = (const float*)d_in[6];
  const float* bx_b = (const float*)d_in[7];
  const float* bh_b = (const float*)d_in[8];
  const float* Wfc  = (const float*)d_in[9];
  const float* bfc  = (const float*)d_in[10];
  float* out = (float*)d_out;

  char* ws = (char*)d_ws;
  __bf16*   whT  = (__bf16*)(ws + OFF_WHT);
  __bf16*   wfcT = (__bf16*)(ws + OFF_WFCT);
  float*    cst  = (float*)(ws + OFF_C);
  unsigned* bar  = (unsigned*)(ws + OFF_BAR);
  __bf16*   hsb  = (__bf16*)(ws + OFF_HS);

  hipMemsetAsync(ws + OFF_BAR, 0, 256, stream);  // capture-legal memset node

  k_prep<<<(4 * NH * NH + 255) / 256, 256, 0, stream>>>(Wh_f, Wh_b, Wfc, whT, wfcT);

  k_bilstm<<<dim3(64, 2), 256, LDS_BYTES, stream>>>(x, Wx_f, bx_f, bh_f, Wx_b, bx_b,
                                                    bh_b, whT, cst, hsb, bar);

  k_fc<<<(NB * NS) / 128, 256, 0, stream>>>(hsb, wfcT, bfc, out);
}
